// MHSA_16956530885053
// MI455X (gfx1250) — matrix operand order repaired and checked by reading, not compiled, not run on hardware
//
#include <hip/hip_runtime.h>
#include <hip/hip_bf16.h>

#define BB 4
#define NN 1024
#define DD 512
#define HH 8
#define DHD 64
#define QK_SCALE 0.125f   // 64^-0.5

typedef __attribute__((ext_vector_type(16))) __bf16 v16bf;
typedef __attribute__((ext_vector_type(8)))  float  v8f;

struct U16x16 { uint4 a, b; };          // 32 bytes
union FragB16 { U16x16 u; v16bf bf; };  // reinterpret raw bf16 bits as WMMA operand

__device__ __forceinline__ unsigned short f2bf(float f) {
  union { float f; unsigned u; } c; c.f = f;
  unsigned r = c.u + 0x7FFFu + ((c.u >> 16) & 1u);   // round-to-nearest-even
  return (unsigned short)(r >> 16);
}

// A-fragment (16x32 bf16, row-major source, leading dim ld in elements).
// ISA layout: lane<16 -> M=lane, K groups {0..7, 16..23}; lane>=16 -> K groups {8..15, 24..31}.
__device__ __forceinline__ v16bf load_a_frag(const unsigned short* p0, int ld) {
  int lane = threadIdx.x & 31;
  int row = lane & 15, half = lane >> 4;
  const unsigned short* p = p0 + (size_t)row * ld + half * 8;
  FragB16 f;
  f.u.a = *(const uint4*)(p);        // K = base+0..7
  f.u.b = *(const uint4*)(p + 16);   // K = base+16..23
  return f.bf;
}

// B-fragment (32x16 bf16). Column n of B is a contiguous length-K run at p0 + n*ld.
// lane<16 -> N=lane, K groups {0..7, 16..23}; lane>=16 -> N=lane-16, K groups {8..15, 24..31}.
__device__ __forceinline__ v16bf load_b_frag(const unsigned short* p0, int ld) {
  int lane = threadIdx.x & 31;
  int col = lane & 15, half = lane >> 4;
  const unsigned short* p = p0 + (size_t)col * ld + half * 8;
  FragB16 f;
  f.u.a = *(const uint4*)(p);
  f.u.b = *(const uint4*)(p + 16);
  return f.bf;
}

__device__ __forceinline__ v8f wmma_bf16(v16bf a, v16bf b, v8f c) {
  return __builtin_amdgcn_wmma_f32_16x16x32_bf16(false, a, false, b, (short)0, c, false, false);
}

__device__ __forceinline__ v8f v8f_zero() {
  v8f z = {0.f,0.f,0.f,0.f,0.f,0.f,0.f,0.f};
  return z;
}

// CDNA5 async copy: global -> LDS, 16 bytes per lane, tracked by ASYNCcnt.
// Generic shared pointers carry the LDS byte offset in their low 32 bits.
__device__ __forceinline__ void async_b128(void* lds_ptr, const void* gptr) {
  unsigned lds_off = (unsigned)(size_t)lds_ptr;
  unsigned long long ga = (unsigned long long)(size_t)gptr;
  asm volatile("global_load_async_to_lds_b128 %0, %1, off"
               :: "v"(lds_off), "v"(ga) : "memory");
}

// ---------------------------------------------------------------- conversion
__global__ void cvt_f32_bf16(const float* __restrict__ src,
                             unsigned short* __restrict__ dst, int n) {
  for (int i = blockIdx.x * blockDim.x + threadIdx.x; i < n;
       i += gridDim.x * blockDim.x)
    dst[i] = f2bf(src[i]);
}

__global__ void init_stats(float* stats) {
  stats[0] = 0.f;                 // entropy accumulator
  ((int*)stats)[1] = 0;           // amax bits (all p > 0 -> positive float bits)
}

// ------------------------------------------------------- QKV projection GEMM
// out[n,o] = sum_k x[n,k] * Wqkv[o,k]; o = s*512 + h*64 + d
// q,k stored [B][H][N][DH] bf16; v stored transposed [B][H][DH][N] bf16.
__launch_bounds__(256)
__global__ void qkv_gemm(const unsigned short* __restrict__ xh,
                         const unsigned short* __restrict__ wh,
                         unsigned short* __restrict__ qh,
                         unsigned short* __restrict__ kh,
                         unsigned short* __restrict__ vt) {
  int wid  = (blockIdx.x * blockDim.x + threadIdx.x) >> 5;
  int lane = threadIdx.x & 31;
  int mtile = wid / 24, ngrp = wid % 24;          // 4096/16=256 M-tiles, 1536/64=24 N-groups
  if (mtile >= 256) return;
  int m0 = mtile * 16, n0 = ngrp * 64;

  v8f acc[4];
  #pragma unroll
  for (int t = 0; t < 4; ++t) acc[t] = v8f_zero();

  for (int k0 = 0; k0 < DD; k0 += 32) {
    v16bf a = load_a_frag(xh + (size_t)m0 * DD + k0, DD);
    #pragma unroll
    for (int t = 0; t < 4; ++t) {
      v16bf b = load_b_frag(wh + (size_t)(n0 + t * 16) * DD + k0, DD);
      acc[t] = wmma_bf16(a, b, acc[t]);
    }
  }

  int colc = lane & 15, half = lane >> 4;
  #pragma unroll
  for (int t = 0; t < 4; ++t) {
    #pragma unroll
    for (int r = 0; r < 8; ++r) {
      int row = m0 + r + 8 * half;               // global token index in [0, B*N)
      int o   = n0 + t * 16 + colc;
      int b   = row / NN, n = row % NN;
      int s   = o / DD;  int rem = o % DD;
      int h   = rem / DHD, d = rem % DHD;
      unsigned short v = f2bf(acc[t][r]);
      size_t head = (size_t)(b * HH + h);
      if (s == 0)      qh[(head * NN + n) * DHD + d] = v;
      else if (s == 1) kh[(head * NN + n) * DHD + d] = v;
      else             vt[(head * DHD + d) * NN + n] = v;
    }
  }
}

// ------------------------------------------------------ fused flash attention
// One wave = 16 query rows of one (b,h). K/V tiles double-buffered in LDS via
// CDNA5 async copies (ASYNCcnt); online softmax + entropy + amax.
__launch_bounds__(64)
__global__ void attn_kernel(const unsigned short* __restrict__ qh,
                            const unsigned short* __restrict__ kh,
                            const unsigned short* __restrict__ vt,
                            const float* __restrict__ bias_table,
                            const int* __restrict__ index,
                            unsigned short* __restrict__ yh,
                            float* __restrict__ stats) {
  __shared__ __align__(16) unsigned short kbuf[2][2][32 * DHD];  // [wave][buf][32 rows x 64 d]
  __shared__ __align__(16) unsigned short vbuf[2][2][DHD * 32];  // [wave][buf][64 d x 32 n]
  __shared__ __align__(16) unsigned short plds[2][16][40];       // P transpose, 80B rows

  int wave = threadIdx.x >> 5, lane = threadIdx.x & 31;
  int colc = lane & 15, half = lane >> 4;

  int tile = blockIdx.x * 2 + wave;                          // 2048 tiles total
  int b   = tile / (HH * (NN / 16));
  int rem = tile % (HH * (NN / 16));
  int h   = rem / (NN / 16);
  int q0  = (rem % (NN / 16)) * 16;

  size_t hb = (size_t)(b * HH + h) * NN * DHD;
  const unsigned short* qb = qh + hb;
  const unsigned short* kb = kh + hb;
  const unsigned short* vb = vt + hb;                        // [DH][N] layout under same base

  // Q fragments resident for the whole sweep (K-dim = DH = 64 -> two 16x32 frags)
  v16bf aq0 = load_a_frag(qb + (size_t)q0 * DHD + 0,  DHD);
  v16bf aq1 = load_a_frag(qb + (size_t)q0 * DHD + 32, DHD);

  float m[8], l[8], tn[8];
  #pragma unroll
  for (int r = 0; r < 8; ++r) { m[r] = -1e30f; l[r] = 0.f; tn[r] = 0.f; }
  v8f acc[4];
  #pragma unroll
  for (int t = 0; t < 4; ++t) acc[t] = v8f_zero();

  // ---- stage j-tile (32 cols of K + V) into LDS buffer `buf` (16 async b128)
  auto stage = [&](int j0, int buf) {
    // K tile: rows j0..j0+31, 64 bf16 each -> one contiguous 4KB block in global
    const char* kg = (const char*)(kb + (size_t)j0 * DHD);
    char* kl = (char*)&kbuf[wave][buf][0];
    #pragma unroll
    for (int it = 0; it < 8; ++it) {
      int c = it * 32 + lane;                                // 16B chunk 0..255
      async_b128(kl + c * 16, kg + c * 16);
    }
    // V tile: 64 rows (d), 32 cols (n=j0..) -> 64B runs, global row stride 2KB
    const char* vg = (const char*)vb + (size_t)j0 * 2;
    char* vl = (char*)&vbuf[wave][buf][0];
    #pragma unroll
    for (int it = 0; it < 8; ++it) {
      int c = it * 32 + lane;                                // 0..255
      int d = c >> 2, sub = c & 3;
      async_b128(vl + d * 64 + sub * 16, vg + (size_t)d * (NN * 2) + sub * 16);
    }
  };

  stage(0, 0);
  int cur = 0;

  for (int j0 = 0; j0 < NN; j0 += 32) {
    if (j0 + 32 < NN) {
      stage(j0 + 32, cur ^ 1);                               // overlap copy with compute
      asm volatile("s_wait_asynccnt 0x10" ::: "memory");     // current stage landed
    } else {
      asm volatile("s_wait_asynccnt 0x0" ::: "memory");
    }
    const unsigned short* kt = &kbuf[wave][cur][0];          // [32 j][64 d]
    const unsigned short* vl = &vbuf[wave][cur][0];          // [64 d][32 n]

    // scores: two 16x16 column tiles, K=64 split into two WMMAs each (ds_load_b128 B-frags)
    v8f s0 = v8f_zero(), s1 = v8f_zero();
    s0 = wmma_bf16(aq0, load_b_frag(kt + 0,              DHD), s0);
    s0 = wmma_bf16(aq1, load_b_frag(kt + 32,             DHD), s0);
    s1 = wmma_bf16(aq0, load_b_frag(kt + 16 * DHD + 0,   DHD), s1);
    s1 = wmma_bf16(aq1, load_b_frag(kt + 16 * DHD + 32,  DHD), s1);

    // scale + relative position bias (gather), then online softmax update
    #pragma unroll
    for (int r = 0; r < 8; ++r) {
      int i = q0 + r + 8 * half;
      s0[r] = s0[r] * QK_SCALE + bias_table[index[i * NN + j0 + colc]      * HH + h];
      s1[r] = s1[r] * QK_SCALE + bias_table[index[i * NN + j0 + 16 + colc] * HH + h];

      float c = fmaxf(s0[r], s1[r]);                         // row max across 16 lanes of half
      c = fmaxf(c, __shfl_xor(c, 1, 32));
      c = fmaxf(c, __shfl_xor(c, 2, 32));
      c = fmaxf(c, __shfl_xor(c, 4, 32));
      c = fmaxf(c, __shfl_xor(c, 8, 32));
      float mn = fmaxf(m[r], c);
      float alpha = __expf(m[r] - mn);
      m[r] = mn;
      l[r] *= alpha; tn[r] *= alpha;
      #pragma unroll
      for (int t = 0; t < 4; ++t) acc[t][r] *= alpha;

      float p0 = __expf(s0[r] - mn);
      float p1 = __expf(s1[r] - mn);
      l[r]  += p0 + p1;
      tn[r] += p0 * s0[r] + p1 * s1[r];                      // sum e^{s-m} * s (entropy)
      plds[wave][r + 8 * half][colc]      = f2bf(p0);        // C-layout -> LDS
      plds[wave][r + 8 * half][16 + colc] = f2bf(p1);
    }
    asm volatile("s_wait_dscnt 0x0" ::: "memory");           // wave-local LDS RAW fence
    v16bf ap = load_a_frag(&plds[wave][0][0], 40);           // read back in A-layout

    #pragma unroll
    for (int t = 0; t < 4; ++t) {                            // P (16x32) x V (32x64)
      v16bf bv = load_b_frag(vl + (t * 16) * 32, 32);
      acc[t] = wmma_bf16(ap, bv, acc[t]);
    }
    cur ^= 1;
  }

  // reduce lane-partial l, t across the 16 lanes of each half
  #pragma unroll
  for (int r = 0; r < 8; ++r) {
    float ls = l[r], ts = tn[r];
    ls += __shfl_xor(ls, 1, 32); ts += __shfl_xor(ts, 1, 32);
    ls += __shfl_xor(ls, 2, 32); ts += __shfl_xor(ts, 2, 32);
    ls += __shfl_xor(ls, 4, 32); ts += __shfl_xor(ts, 4, 32);
    ls += __shfl_xor(ls, 8, 32); ts += __shfl_xor(ts, 8, 32);
    l[r] = ls; tn[r] = ts;
  }

  // normalize and store output heads as bf16 [B][N][D]
  size_t yb = (size_t)b * NN * DD;
  #pragma unroll
  for (int t = 0; t < 4; ++t)
    #pragma unroll
    for (int r = 0; r < 8; ++r) {
      int n = q0 + r + 8 * half;
      yh[yb + (size_t)n * DD + h * DHD + t * 16 + colc] = f2bf(acc[t][r] / l[r]);
    }

  // stats: ent_row = m + log(l) - t/l ; pmax_row = 1/l (row max logit == m)
  if (colc == 0) {
    float ent = 0.f, pmax = 0.f;
    #pragma unroll
    for (int r = 0; r < 8; ++r) {
      ent += m[r] + __logf(l[r]) - tn[r] / l[r];
      pmax = fmaxf(pmax, 1.0f / l[r]);
    }
    atomicAdd(&stats[0], ent);
    atomicMax((int*)stats + 1, __float_as_int(pmax));
  }
}

// ---------------------------------------------------------- output projection
__launch_bounds__(256)
__global__ void proj_gemm(const unsigned short* __restrict__ yh,
                          const unsigned short* __restrict__ wh,
                          float* __restrict__ out) {
  int wid  = (blockIdx.x * blockDim.x + threadIdx.x) >> 5;
  int lane = threadIdx.x & 31;
  int mtile = wid / 8, ngrp = wid % 8;            // 256 M-tiles x 8 N-groups
  if (mtile >= 256) return;
  int m0 = mtile * 16, n0 = ngrp * 64;

  v8f acc[4];
  #pragma unroll
  for (int t = 0; t < 4; ++t) acc[t] = v8f_zero();

  for (int k0 = 0; k0 < DD; k0 += 32) {
    v16bf a = load_a_frag(yh + (size_t)m0 * DD + k0, DD);
    #pragma unroll
    for (int t = 0; t < 4; ++t) {
      v16bf b = load_b_frag(wh + (size_t)(n0 + t * 16) * DD + k0, DD);
      acc[t] = wmma_bf16(a, b, acc[t]);
    }
  }

  int colc = lane & 15, half = lane >> 4;
  #pragma unroll
  for (int t = 0; t < 4; ++t)
    #pragma unroll
    for (int r = 0; r < 8; ++r) {
      int row = m0 + r + 8 * half;
      out[(size_t)row * DD + n0 + t * 16 + colc] = acc[t][r];
    }
}

__global__ void finalize_stats(const float* __restrict__ stats, float* __restrict__ tail) {
  tail[0] = stats[0] / (float)(BB * HH * NN);     // ent mean over rows
  tail[1] = __int_as_float(((const int*)stats)[1]);
  tail[2] = 1.0f / (float)NN;                     // softmax rows sum to 1 -> mean = 1/N
}

// ------------------------------------------------------------------- launcher
extern "C" void kernel_launch(void* const* d_in, const int* in_sizes, int n_in,
                              void* d_out, int out_size, void* d_ws, size_t ws_size,
                              hipStream_t stream) {
  (void)in_sizes; (void)n_in; (void)out_size; (void)ws_size;
  const float* x     = (const float*)d_in[0];
  const float* Wqkv  = (const float*)d_in[1];
  const float* Wproj = (const float*)d_in[2];
  const float* btab  = (const float*)d_in[3];
  const int*   index = (const int*)d_in[4];
  float* out = (float*)d_out;

  const size_t XN  = (size_t)BB * NN * DD;         // 2,097,152
  const size_t WQN = (size_t)3 * DD * DD;          //   786,432
  const size_t WPN = (size_t)DD * DD;              //   262,144
  const size_t HQN = (size_t)BB * HH * NN * DHD;   // 2,097,152

  char* w = (char*)d_ws;
  float*          stats  = (float*)w;          w += 256;
  unsigned short* xh     = (unsigned short*)w; w += XN  * 2;
  unsigned short* wqkvh  = (unsigned short*)w; w += WQN * 2;
  unsigned short* wprojh = (unsigned short*)w; w += WPN * 2;
  unsigned short* qh     = (unsigned short*)w; w += HQN * 2;
  unsigned short* kh     = (unsigned short*)w; w += HQN * 2;
  unsigned short* vt     = (unsigned short*)w; w += HQN * 2;
  unsigned short* yh     = (unsigned short*)w; w += XN  * 2;

  init_stats<<<1, 1, 0, stream>>>(stats);
  cvt_f32_bf16<<<2048, 256, 0, stream>>>(x,     xh,     (int)XN);
  cvt_f32_bf16<<<1024, 256, 0, stream>>>(Wqkv,  wqkvh,  (int)WQN);
  cvt_f32_bf16<<<512,  256, 0, stream>>>(Wproj, wprojh, (int)WPN);

  // 256 M-tiles * 24 N-groups = 6144 waves, 8 waves/block
  qkv_gemm<<<768, 256, 0, stream>>>(xh, wqkvh, qh, kh, vt);

  // B*H*(N/16) = 2048 wave-tiles, 2 waves/block (LDS double buffers)
  attn_kernel<<<1024, 64, 0, stream>>>(qh, kh, vt, btab, index, yh, stats);

  // 256 M-tiles * 8 N-groups = 2048 waves, 8 waves/block
  proj_gemm<<<256, 256, 0, stream>>>(yh, wprojh, out);

  finalize_stats<<<1, 1, 0, stream>>>(stats, out + XN);
}
